// EuclideanToLorentzConv_1554778161716
// MI455X (gfx1250) — compile-verified, matrix operand order checked by reading, not run on hardware
//
#include <hip/hip_runtime.h>
#include <hip/hip_bf16.h>
#include <stdint.h>

typedef __attribute__((ext_vector_type(16))) _Float16 v16h;
typedef __attribute__((ext_vector_type(8)))  float    v8f;

#define BATCH 16
#define HH    112
#define WW    112
#define NPIX  (HH*WW)          // 12544
#define TOT   (BATCH*NPIX)     // 200704
#define KCH   18               // 576/32 k-chunks
#define GEMM_THREADS 224       // 7 waves of 32

union FragH { v16h v; uint32_t u[8]; uint4 q[2]; };

__device__ __forceinline__ uint32_t pack16(_Float16 a, _Float16 b) {
  union { _Float16 h[2]; uint32_t u; } t;
  t.h[0] = a; t.h[1] = b;
  return t.u;
}
// half2(hi,lo) split of an f32: hi in bits[15:0], lo in bits[31:16]
__device__ __forceinline__ uint32_t pack_hl(float f) {
  _Float16 h = (_Float16)f;
  _Float16 l = (_Float16)(f - (float)h);
  return pack16(h, l);
}

__device__ __forceinline__ float blockReduceSum(float v) {
  #pragma unroll
  for (int o = 16; o > 0; o >>= 1) v += __shfl_xor(v, o, 32);
  __shared__ float sm[8];
  int lane = threadIdx.x & 31, wv = threadIdx.x >> 5;
  if (lane == 0) sm[wv] = v;
  __syncthreads();
  if (wv == 0) {
    v = (lane < 8) ? sm[lane] : 0.f;
    #pragma unroll
    for (int o = 4; o > 0; o >>= 1) v += __shfl_xor(v, o, 32);
  }
  return v; // valid on wave0
}

// -------------------- small prep kernels --------------------
__global__ void k_zero(float* musum, float* varsum) {
  int t = threadIdx.x;
  if (t < 128) musum[t] = 0.f;
  if (t == 0)  varsum[0] = 0.f;
}

// t2[b,h,w] = 1 + sum_{c=1..63} x^2   (== t^2)
__global__ void k_t2(const float* __restrict__ x, float* __restrict__ t2) {
  int pid = blockIdx.x * 256 + threadIdx.x;
  if (pid >= TOT) return;
  int b = pid / NPIX, p = pid % NPIX;
  const float* xp = x + (size_t)b * 64 * NPIX + p;
  float s = 1.f;
  #pragma unroll 7
  for (int c = 1; c < 64; ++c) { float v = xp[(size_t)c * NPIX]; s += v * v; }
  t2[pid] = s;
}

// T = sqrt(max(boxsum3x3(t2, pad=1) - 8, eps))
__global__ void k_T(const float* __restrict__ t2, float* __restrict__ T) {
  int pid = blockIdx.x * 256 + threadIdx.x;
  if (pid >= TOT) return;
  int b = pid / NPIX, p = pid % NPIX;
  int h = p / WW, w = p % WW;
  float s = 0.f;
  #pragma unroll
  for (int di = -1; di <= 1; ++di)
    #pragma unroll
    for (int dj = -1; dj <= 1; ++dj) {
      int hh = h + di, ww = w + dj;
      s += (hh >= 0 && hh < HH && ww >= 0 && ww < WW)
             ? t2[b * NPIX + hh * WW + ww] : 1.f;
    }
  T[pid] = sqrtf(fmaxf(s - 8.f, 1e-6f));
}

// Pre-swizzle W into A-fragment layout: Wp[ch][prec][mt][lane][8 words],
// word = pack(f16(k), f16(k+1)) of hi or lo part.  k' = tap*64 + c.
__device__ __forceinline__ float wraw(const float* Wt, int m, int k) {
  if (m >= 127) return 0.f;
  int tap = k >> 6, c = k & 63;
  if (c < 63) return Wt[(size_t)m * 568 + 1 + tap * 63 + c];
  return (tap == 0) ? Wt[(size_t)m * 568] : 0.f; // T column in slot (tap0,c63)
}
__global__ void k_packW(const float* __restrict__ Wt, uint32_t* __restrict__ Wp) {
  int u = blockIdx.x * 256 + threadIdx.x;
  if (u >= KCH * 4096) return;
  int w8   = u & 7;
  int L    = (u >> 3) & 31;
  int mt   = (u >> 8) & 7;
  int prec = (u >> 11) & 1;
  int ch   = u >> 12;
  int m  = mt * 16 + (L & 15);
  int kl = ((w8 < 4) ? 2 * w8 : 16 + 2 * (w8 - 4)) + ((L >> 4) * 8);
  int k0 = ch * 32 + kl;
  float a = wraw(Wt, m, k0), b = wraw(Wt, m, k0 + 1);
  _Float16 ha, hb;
  if (prec == 0) { ha = (_Float16)a; hb = (_Float16)b; }
  else {
    _Float16 t0 = (_Float16)a, t1 = (_Float16)b;
    ha = (_Float16)(a - (float)t0); hb = (_Float16)(b - (float)t1);
  }
  Wp[u] = pack16(ha, hb);
}

// -------------------- WMMA implicit-GEMM conv --------------------
__device__ __forceinline__ uint32_t fetchB(int c, int tap, int rj, int pl,
                                           const uint32_t* sx, const uint32_t* sT) {
  if (c < 63) return sx[c * 114 + pl + rj];
  return (tap == 0) ? sT[pl] : 0u;
}

__global__ __launch_bounds__(GEMM_THREADS)
void k_gemm(const float* __restrict__ x, const uint32_t* __restrict__ Wp,
            const float* __restrict__ Tg, float* __restrict__ out) {
  __shared__ __align__(16) uint32_t sx[63 * 114];  // half2 row buffer (one h-row of s)
  __shared__ __align__(16) uint32_t sA[2 * 4096];  // double-buffered W chunks
  __shared__ __align__(16) uint32_t sT[114];       // T half2 row
  const int tid  = threadIdx.x;
  const int lane = tid & 31;
  const int wv   = tid >> 5;
  const int b = blockIdx.x / HH, h = blockIdx.x % HH;
  const int n = lane & 15;
  const int plocal = wv * 16 + n;        // pixel (w) 0..111
  const int basek  = (lane >> 4) * 8;
  // LDS byte offset of sA (generic LDS pointer carries offset in low 32 bits)
  const uint32_t sAoff = (uint32_t)(uintptr_t)(void*)sA;

  v8f acc[8];
  #pragma unroll
  for (int i = 0; i < 8; ++i) acc[i] = (v8f){0.f,0.f,0.f,0.f,0.f,0.f,0.f,0.f};

  for (int i = tid; i < 114; i += GEMM_THREADS)
    sT[i] = (i < WW) ? pack_hl(Tg[b * NPIX + h * WW + i]) : 0u;

  // async preload of chunk 0 into buffer 0 (waves 0-3, 8 x b128 each)
  if (wv < 4) {
    #pragma unroll
    for (int it = 0; it < 8; ++it) {
      int i = (wv * 8 + it) * 32 + lane;
      uint32_t loff = sAoff + (uint32_t)i * 16u;
      const uint32_t* g = Wp + (size_t)i * 4;
      asm volatile("global_load_async_to_lds_b128 %0, %1, off"
                   :: "v"(loff), "v"(g) : "memory");
    }
  }

  #pragma unroll
  for (int ch = 0; ch < KCH; ++ch) {
    const int tap   = ch >> 1;
    const int ri    = tap / 3, rj = tap % 3;
    const int cbase = (ch & 1) * 32;
    const int buf   = ch & 1;
    __syncthreads();  // everyone done with previous chunk; buf^1 is free
    if ((ch % 6) == 0) {                 // stage new s-row (shift ri-1)
      int hy = h + ri - 1;
      for (int idx = tid; idx < 63 * 114; idx += GEMM_THREADS) {
        int c = idx / 114, cc = idx % 114, wg = cc - 1;
        float xv = 0.f;
        if (hy >= 0 && hy < HH && wg >= 0 && wg < WW)
          xv = x[(((size_t)b * 64 + (c + 1)) * HH + hy) * WW + wg];
        sx[idx] = pack_hl(xv);
      }
    }
    // prefetch next W chunk into the other buffer (async, no VGPR traffic)
    if (wv < 4) {
      if (ch + 1 < KCH) {
        #pragma unroll
        for (int it = 0; it < 8; ++it) {
          int i = (wv * 8 + it) * 32 + lane;
          uint32_t loff = sAoff + (uint32_t)(buf ^ 1) * 16384u + (uint32_t)i * 16u;
          const uint32_t* g = Wp + (size_t)(ch + 1) * 4096 + (size_t)i * 4;
          asm volatile("global_load_async_to_lds_b128 %0, %1, off"
                       :: "v"(loff), "v"(g) : "memory");
        }
        asm volatile("s_wait_asynccnt 0x8" ::: "memory"); // chunk ch has landed
      } else {
        asm volatile("s_wait_asynccnt 0x0" ::: "memory"); // last chunk landed
      }
    }
    __syncthreads();  // publish LDS (async W chunk + s-row) to all waves

    // build B fragments (K x N = 32x16), hi & lo, from half2 LDS reads
    FragH bhf, blf;
    #pragma unroll
    for (int w8 = 0; w8 < 8; ++w8) {
      int kl = ((w8 < 4) ? 2 * w8 : 16 + 2 * (w8 - 4)) + basek;
      uint32_t u0 = fetchB(cbase + kl,     tap, rj, plocal, sx, sT);
      uint32_t u1 = fetchB(cbase + kl + 1, tap, rj, plocal, sx, sT);
      bhf.u[w8] = (u0 & 0xFFFFu) | (u1 << 16);
      blf.u[w8] = (u0 >> 16)     | (u1 & 0xFFFF0000u);
    }

    // 8 M-tiles x (hi*hi + hi*lo + lo*hi)
    const uint32_t* sAb = sA + buf * 4096;
    #pragma unroll
    for (int mt = 0; mt < 8; ++mt) {
      FragH ah, al;
      const uint32_t* pa = sAb + mt * 256 + lane * 8;
      ah.q[0] = *(const uint4*)pa;            ah.q[1] = *(const uint4*)(pa + 4);
      const uint32_t* pl = sAb + 2048 + mt * 256 + lane * 8;
      al.q[0] = *(const uint4*)pl;            al.q[1] = *(const uint4*)(pl + 4);
      acc[mt] = __builtin_amdgcn_wmma_f32_16x16x32_f16(false, ah.v, false, bhf.v,
                                                       (short)0, acc[mt], false, false);
      acc[mt] = __builtin_amdgcn_wmma_f32_16x16x32_f16(false, ah.v, false, blf.v,
                                                       (short)0, acc[mt], false, false);
      acc[mt] = __builtin_amdgcn_wmma_f32_16x16x32_f16(false, al.v, false, bhf.v,
                                                       (short)0, acc[mt], false, false);
    }
  }

  // epilogue: y_t = sqrt(1 + sum y_s^2), write y = [y_t, y_s]
  float ysum = 0.f;
  #pragma unroll
  for (int mt = 0; mt < 8; ++mt)
    #pragma unroll
    for (int r = 0; r < 8; ++r) { float v = acc[mt][r]; ysum += v * v; }
  ysum += __shfl_xor(ysum, 16, 32);
  float yt = sqrtf(1.f + ysum);

  size_t pixoff = (size_t)h * WW + plocal;
  size_t bbase  = (size_t)b * 128 * NPIX;
  if (lane < 16) out[bbase + pixoff] = yt;
  const int mofs = (lane >> 4) * 8;
  #pragma unroll
  for (int mt = 0; mt < 8; ++mt)
    #pragma unroll
    for (int r = 0; r < 8; ++r) {
      int M = mt * 16 + r + mofs;
      if (M < 127) out[bbase + (size_t)(M + 1) * NPIX + pixoff] = acc[mt][r];
    }
}

// -------------------- Lorentz batch-norm pipeline --------------------
__global__ void k_musum(const float* __restrict__ y, float* __restrict__ musum) {
  int c = blockIdx.y;
  float s = 0.f;
  for (int idx = blockIdx.x * 256 + threadIdx.x; idx < TOT; idx += gridDim.x * 256) {
    int b = idx / NPIX, p = idx % NPIX;
    s += y[((size_t)b * 128 + c) * NPIX + p];
  }
  s = blockReduceSum(s);
  if (threadIdx.x == 0) atomicAdd(&musum[c], s);
}

__global__ void k_mufin(const float* __restrict__ musum, float* __restrict__ mu) {
  int c = threadIdx.x; // 128 threads
  float m = musum[c] / (float)TOT;
  __shared__ float sm[128];
  sm[c] = (c == 0) ? 0.f : m * m;
  __syncthreads();
  __shared__ float denom;
  if (c == 0) {
    float s = 0.f;
    for (int i = 1; i < 128; ++i) s += sm[i];
    float m0 = musum[0] / (float)TOT;
    denom = sqrtf(m0 * m0 - s);
  }
  __syncthreads();
  float mv = m / denom;
  mu[c] = mv;
  if (c == 0) mu[128] = 1.f / (1.f + mv);
}

__global__ void k_v(float* __restrict__ y, const float* __restrict__ mu,
                    float* __restrict__ varsum) {
  __shared__ float smu[129];
  if (threadIdx.x < 129) smu[threadIdx.x] = mu[threadIdx.x];
  __syncthreads();
  int pid = blockIdx.x * 256 + threadIdx.x;
  float q = 0.f;
  if (pid < TOT) {
    int b = pid / NPIX, p = pid % NPIX;
    float* yp = y + (size_t)b * 128 * NPIX + p;
    float mu0 = smu[0];
    float y0 = yp[0];
    float a = mu0 * y0;
    for (int c = 1; c < 128; ++c) a -= smu[c] * yp[(size_t)c * NPIX];
    float alpha = fmaxf(a, 1.f + 1e-6f);
    float den  = sqrtf(alpha * alpha - 1.f);
    float coef = acoshf(alpha) / den;
    float v0   = coef * (y0 - alpha * mu0);
    float proj = v0 * smu[128];            // v0 / (1+mu0)
    float vz = v0 - proj * (mu0 + 1.f);
    yp[0] = vz;
    q = -vz * vz;
    for (int c = 1; c < 128; ++c) {
      float vc = coef * (yp[(size_t)c * NPIX] - alpha * smu[c]) - proj * smu[c];
      yp[(size_t)c * NPIX] = vc;
      q += vc * vc;
    }
  }
  float s = blockReduceSum(q);
  if (threadIdx.x == 0) atomicAdd(varsum, s);
}

__global__ void k_varfin(const float* varsum, const float* gamma, float* scale) {
  float var = varsum[0] / (float)TOT;
  scale[0] = gamma[0] / sqrtf(var + 1e-5f);
}

__global__ void k_final(float* __restrict__ y, const float* __restrict__ beta,
                        const float* __restrict__ scale) {
  __shared__ float sb[128];
  __shared__ float ssc;
  if (threadIdx.x < 128) sb[threadIdx.x] = beta[threadIdx.x];
  if (threadIdx.x == 0)  ssc = scale[0];
  __syncthreads();
  int pid = blockIdx.x * 256 + threadIdx.x;
  if (pid >= TOT) return;
  int b = pid / NPIX, p = pid % NPIX;
  float* yp = y + (size_t)b * 128 * NPIX + p;
  float sc = ssc, b0 = sb[0];
  float vs0 = sc * yp[0];
  float bd = -b0 * vs0;
  for (int c = 1; c < 128; ++c) bd += sb[c] * sc * yp[(size_t)c * NPIX];
  float inv = bd / (1.f + b0);
  float vb0 = vs0 + inv * (1.f + b0);
  float q = -vb0 * vb0;
  for (int c = 1; c < 128; ++c) {
    float vb = sc * yp[(size_t)c * NPIX] + inv * sb[c];
    q += vb * vb;
  }
  float vn  = sqrtf(fmaxf(q, 1e-6f));
  float chv = coshf(vn), shv = sinhf(vn) / vn;
  float rsum = 0.f;
  for (int c = 1; c < 128; ++c) {
    float vb = sc * yp[(size_t)c * NPIX] + inv * sb[c];
    float z  = chv * sb[c] + shv * vb;
    float rs = fmaxf(z, 0.f);
    rsum += rs * rs;
    yp[(size_t)c * NPIX] = rs;
  }
  yp[0] = sqrtf(1.f + rsum);
}

// -------------------- launch --------------------
extern "C" void kernel_launch(void* const* d_in, const int* in_sizes, int n_in,
                              void* d_out, int out_size, void* d_ws, size_t ws_size,
                              hipStream_t stream) {
  const float* x     = (const float*)d_in[0];
  const float* Wt    = (const float*)d_in[1];
  const float* gamma = (const float*)d_in[2];
  const float* beta  = (const float*)d_in[3];
  float* out = (float*)d_out;
  float* ws  = (float*)d_ws;

  float*    t2     = ws;                         // 200704 f
  float*    Tbuf   = ws + TOT;                   // 200704 f
  uint32_t* Wp     = (uint32_t*)(ws + 2 * TOT);  // 73728 u32 (16B aligned)
  float*    musum  = ws + 2 * TOT + KCH * 4096;  // 128 f
  float*    varsum = musum + 128;                // 1 f
  float*    mu     = varsum + 1;                 // 129 f (mu[128] + 1/(1+mu0))
  float*    scale  = mu + 130;                   // 1 f

  k_zero <<<1, 256, 0, stream>>>(musum, varsum);
  k_packW<<<(KCH * 4096 + 255) / 256, 256, 0, stream>>>(Wt, Wp);
  k_t2   <<<(TOT + 255) / 256, 256, 0, stream>>>(x, t2);
  k_T    <<<(TOT + 255) / 256, 256, 0, stream>>>(t2, Tbuf);
  k_gemm <<<BATCH * HH, GEMM_THREADS, 0, stream>>>(x, Wp, Tbuf, out);
  dim3 g(98, 128);
  k_musum<<<g, 256, 0, stream>>>(out, musum);
  k_mufin<<<1, 128, 0, stream>>>(musum, mu);
  k_v    <<<(TOT + 255) / 256, 256, 0, stream>>>(out, mu, varsum);
  k_varfin<<<1, 1, 0, stream>>>(varsum, gamma, scale);
  k_final<<<(TOT + 255) / 256, 256, 0, stream>>>(out, beta, scale);
}
